// HardNegativeContrastiveLoss_11046655885427
// MI455X (gfx1250) — matrix operand order repaired and checked by reading, not compiled, not run on hardware
//
#include <hip/hip_runtime.h>
#include <math.h>

// ---------------------------------------------------------------------------
// HardNegativeContrastiveLoss for MI455X (gfx1250, wave32, WMMA)
//   B = 8192 rows, D = 512 features, 64 labels
//   ws layout:  [0, 8MB)   : f16 normalized features (8192*512*2 bytes)
//               [8MB, ...) : candidate indices int32 [B][9] (slot0=pos, 1..8=neg)
// ---------------------------------------------------------------------------

typedef __attribute__((ext_vector_type(16))) _Float16 v16h;
typedef __attribute__((ext_vector_type(8)))  float    v8f;

#define BDIM   8192
#define DDIM   512
#define TEMP_INV 2.0f      // 1 / 0.5
#define NNEG   8
#define HARDK  3

// ---- deterministic counter-based "gumbel" (stand-in for jax threefry) -----
__device__ __forceinline__ float gumbel_hash(unsigned seed, unsigned i, unsigned j) {
    unsigned h = seed;
    h ^= i * 0x9E3779B1u;  h = (h ^ (h >> 16)) * 0x85EBCA6Bu;
    h ^= j * 0xC2B2AE35u;  h = (h ^ (h >> 13)) * 0x27D4EB2Fu;
    h ^= h >> 16;
    // uniform in (0,1), then Gumbel(0,1)
    float u = (float)(h >> 8) * (1.0f / 16777216.0f) + (0.5f / 16777216.0f);
    return -__logf(-__logf(u));
}

// ---------------------------------------------------------------------------
// Kernel 0: zero the scalar output (graph-replay safe)
// ---------------------------------------------------------------------------
__global__ void k_zero(float* out) {
    if (threadIdx.x == 0) out[0] = 0.0f;
}

// ---------------------------------------------------------------------------
// Kernel 1: L2-normalize rows, emit fp16 copy for the WMMA pass.
//   grid = B blocks, 128 threads; each thread owns 4 elements of the row.
// ---------------------------------------------------------------------------
__global__ void k_normalize(const float* __restrict__ feat,
                            _Float16* __restrict__ f16out) {
    __shared__ float red[128];
    const int row = blockIdx.x;
    const int tid = threadIdx.x;
    const float* src = feat + (size_t)row * DDIM;

    float v[4];
    float ss = 0.0f;
#pragma unroll
    for (int e = 0; e < 4; ++e) {
        v[e] = src[tid + e * 128];
        ss += v[e] * v[e];
    }
    red[tid] = ss;
    __syncthreads();
    for (int s = 64; s > 0; s >>= 1) {
        if (tid < s) red[tid] += red[tid + s];
        __syncthreads();
    }
    const float inv = 1.0f / fmaxf(sqrtf(red[0]), 1e-12f);
    _Float16* dst = f16out + (size_t)row * DDIM;
#pragma unroll
    for (int e = 0; e < 4; ++e)
        dst[tid + e * 128] = (_Float16)(v[e] * inv);
}

// ---------------------------------------------------------------------------
// Kernel 2: sample 1 positive (argmax gumbel over same-label) and 8 negatives
// (top-8 gumbel over different-label) per row.  grid = B blocks x 256 thr.
// ---------------------------------------------------------------------------
__global__ void k_sample(const long long* __restrict__ labels,
                         int* __restrict__ cand) {
    __shared__ float sPv[256];
    __shared__ int   sPi[256];
    __shared__ float sNv[256][NNEG];
    __shared__ int   sNi[256][NNEG];

    const int row = blockIdx.x;
    const int tid = threadIdx.x;
    const long long mylab = labels[row];

    float pv = -INFINITY; int pi = -1;
    float nv[NNEG]; int ni[NNEG];
#pragma unroll
    for (int k = 0; k < NNEG; ++k) { nv[k] = -INFINITY; ni[k] = -1; }

    for (int j = tid; j < BDIM; j += 256) {
        const bool same = (labels[j] == mylab);
        if (same) {
            if (j != row) {
                const float g = gumbel_hash(0x1234567u, row, j);
                if (g > pv) { pv = g; pi = j; }
            }
        } else {
            const float g = gumbel_hash(0x89ABCDEFu, row, j);
            if (g > nv[NNEG - 1]) {           // sorted-descending insertion
                int k = NNEG - 1;
                while (k > 0 && g > nv[k - 1]) {
                    nv[k] = nv[k - 1]; ni[k] = ni[k - 1]; --k;
                }
                nv[k] = g; ni[k] = j;
            }
        }
    }

    sPv[tid] = pv; sPi[tid] = pi;
#pragma unroll
    for (int k = 0; k < NNEG; ++k) { sNv[tid][k] = nv[k]; sNi[tid][k] = ni[k]; }
    __syncthreads();

    // tree reduce: argmax for positive, sorted top-8 merge for negatives
    for (int s = 128; s > 0; s >>= 1) {
        if (tid < s) {
            if (sPv[tid + s] > sPv[tid]) {
                sPv[tid] = sPv[tid + s]; sPi[tid] = sPi[tid + s];
            }
            float mv[NNEG]; int mi[NNEG];
            int pa = 0, pb = 0;
#pragma unroll
            for (int k = 0; k < NNEG; ++k) {
                const float va = sNv[tid][pa], vb = sNv[tid + s][pb];
                if (va >= vb) { mv[k] = va; mi[k] = sNi[tid][pa]; ++pa; }
                else          { mv[k] = vb; mi[k] = sNi[tid + s][pb]; ++pb; }
            }
#pragma unroll
            for (int k = 0; k < NNEG; ++k) { sNv[tid][k] = mv[k]; sNi[tid][k] = mi[k]; }
        }
        __syncthreads();
    }

    if (tid == 0) {
        int* c = cand + (size_t)row * 9;
        c[0] = (sPi[0] >= 0) ? sPi[0] : row;   // degenerate: self (sim=1)
#pragma unroll
        for (int k = 0; k < NNEG; ++k) c[1 + k] = (sNi[0][k] >= 0) ? sNi[0][k] : row;
    }
}

// ---------------------------------------------------------------------------
// Kernel 3: WMMA similarity + loss.  One wave (32 thr) per block of 16 rows.
//   A = f16 rows b0..b0+15  (16 x 32 f16 fragment per K step)
//   B = 144 gathered candidate columns, 9 tiles of 16
//   Loop order: K outermost, 9 accumulator tiles innermost.  A fragment is
//   loaded once per K step and reused by 9 WMMAs; live state stays ~110 VGPRs
//   (9x8 f32 acc + 8 A + 8 B + pointers) so nothing spills to scratch.
// ---------------------------------------------------------------------------
__global__ void k_wmma_loss(const _Float16* __restrict__ f16,
                            const int* __restrict__ cand,
                            float* __restrict__ out) {
    __shared__ int   sCand[16 * 9];     // 144 candidate row indices
    __shared__ float sSims[16][9];

    const int lane = threadIdx.x;       // 0..31, one wave
    const int b0   = blockIdx.x * 16;

    for (int c = lane; c < 144; c += 32)
        sCand[c] = cand[(size_t)b0 * 9 + c];
    __syncthreads();

    // A fragment addressing (16-bit A 16x32 layout):
    //   lane L: row M = L&15; v0..v3 hold K = kbase..kbase+7 (kbase = (L>>4)*8),
    //           v4..v7 hold K = 16+kbase..16+kbase+7
    const int Mrow  = lane & 15;
    const int khalf = (lane >> 4);                       // 0 or 1
    const _Float16* rowA = f16 + (size_t)(b0 + Mrow) * DDIM + khalf * 8;

    // B fragment (16-bit B 32x16): lane L: col N = L&15,
    //   same K layout as A: v0..v3 hold K = kbase..kbase+7, v4..v7 hold K = 16+kbase..16+kbase+7
    const _Float16* rowB[9];
#pragma unroll
    for (int t = 0; t < 9; ++t)
        rowB[t] = f16 + (size_t)sCand[t * 16 + Mrow] * DDIM + khalf * 8;

    v8f acc[9];
#pragma unroll
    for (int t = 0; t < 9; ++t) acc[t] = (v8f){};

#pragma unroll 1
    for (int kk = 0; kk < DDIM / 32; ++kk) {
        v16h a;
#pragma unroll
        for (int e = 0; e < 8; ++e) {
            a[e]     = rowA[kk * 32 + e];
            a[e + 8] = rowA[kk * 32 + 16 + e];
        }
#pragma unroll
        for (int t = 0; t < 9; ++t) {
            v16h b;
#pragma unroll
            for (int e = 0; e < 16; ++e)
                b[e] = rowB[t][kk * 32 + (e < 8 ? e : 8 + e)];
            acc[t] = __builtin_amdgcn_wmma_f32_16x16x32_f16(
                         false, a, false, b, (short)0, acc[t], false, false);
        }
    }

    // C layout: lane L holds column N=L&15; VGPR r holds row M=r+8*(L>>4).
#pragma unroll
    for (int t = 0; t < 9; ++t) {
        const int cg = t * 16 + Mrow;          // candidate column 0..143
#pragma unroll
        for (int r = 0; r < 8; ++r) {
            const int M = r + 8 * khalf;
            if (cg / 9 == M)                   // this entry is row M's slot
                sSims[M][cg - M * 9] = acc[t][r];
        }
    }
    __syncthreads();

    if (lane < 16) {
        const float* s = sSims[lane];
        const float pos = s[0] * TEMP_INV;
        float t1 = -INFINITY, t2 = -INFINITY, t3 = -INFINITY;
#pragma unroll
        for (int m = 1; m <= NNEG; ++m) {
            const float x = s[m];
            if      (x > t1) { t3 = t2; t2 = t1; t1 = x; }
            else if (x > t2) { t3 = t2; t2 = x; }
            else if (x > t3) { t3 = x; }
        }
        const float l1 = t1 * TEMP_INV, l2 = t2 * TEMP_INV, l3 = t3 * TEMP_INV;
        const float mx = fmaxf(fmaxf(pos, l1), fmaxf(l2, l3));
        const float lse = mx + __logf(__expf(pos - mx) + __expf(l1 - mx) +
                                      __expf(l2 - mx) + __expf(l3 - mx));
        atomicAdd(out, (lse - pos) * (1.0f / (float)BDIM));
    }
}

// ---------------------------------------------------------------------------
extern "C" void kernel_launch(void* const* d_in, const int* in_sizes, int n_in,
                              void* d_out, int out_size, void* d_ws, size_t ws_size,
                              hipStream_t stream) {
    const float*     feat   = (const float*)d_in[0];
    const long long* labels = (const long long*)d_in[1];
    float*           out    = (float*)d_out;

    _Float16* f16feat = (_Float16*)d_ws;                       // 8 MB
    int*      cand    = (int*)((char*)d_ws + (size_t)BDIM * DDIM * 2);

    k_zero<<<1, 32, 0, stream>>>(out);
    k_normalize<<<BDIM, 128, 0, stream>>>(feat, f16feat);
    k_sample<<<BDIM, 256, 0, stream>>>(labels, cand);
    k_wmma_loss<<<BDIM / 16, 32, 0, stream>>>(f16feat, cand, out);
}
